// DecoderRNN_8667244003660
// MI455X (gfx1250) — matrix operand order repaired and checked by reading, not compiled, not run on hardware
//
#include <hip/hip_runtime.h>
#include <hip/hip_bf16.h>

// Problem constants (from reference)
#define B_     32
#define N_     49
#define ENC_   2048
#define E_     512
#define ATT_   512
#define H_     512
#define V_     30522
#define T_     21
#define STEPS_ 20
#define KCOMB_ 3072   // E + ENC + H  (fused [x | context | h] width)

typedef __bf16 bf16_t;
typedef __attribute__((ext_vector_type(16))) __bf16 bf16x16;
typedef __attribute__((ext_vector_type(8)))  __bf16 bf16x8;
typedef __attribute__((ext_vector_type(8)))  float  f32x8;

// ---------------------------------------------------------------------------
// Generic bf16 GEMM:  C[M x N] = A[M x K] @ B[N x K]^T + bias[N]
// One wave per 32x16 output block (two stacked 16x16 WMMA tiles): each B
// fragment is loaded once and feeds two v_wmma_f32_16x16x32_bf16, halving
// B-matrix traffic (the roofline term for the 31.3MB vocab weight).
// Requires: K % 32 == 0, M % 32 == 0 (true for all call sites), N arbitrary.
// ---------------------------------------------------------------------------
__global__ void __launch_bounds__(256)
gemm_bf16_wmma(const __hip_bfloat16* __restrict__ Ah,
               const __hip_bfloat16* __restrict__ Bh,
               float* __restrict__ C,
               const float* __restrict__ bias,
               int M, int N, int K, long ldc)
{
    const bf16_t* A  = (const bf16_t*)Ah;
    const bf16_t* Bm = (const bf16_t*)Bh;

    int mPairs = M >> 5;                         // 32-row blocks
    int nTiles = (N + 15) >> 4;
    int wid = (blockIdx.x * blockDim.x + threadIdx.x) >> 5;
    if (wid >= mPairs * nTiles) return;          // uniform per wave

    int tn   = wid % nTiles;
    int tp   = wid / nTiles;
    int lane = threadIdx.x & 31;
    int half = lane >> 4;                        // 0: lanes 0-15, 1: lanes 16-31
    int l15  = lane & 15;

    int m0 = tp * 32 + l15;                      // A row, upper 16x16 tile
    int n  = tn * 16 + l15;                      // output column handled by this lane
    bool nOK = (n < N);

    const bf16_t* arow0 = A + (long)m0 * K;
    const bf16_t* arow1 = arow0 + (long)16 * K;
    // Clamp OOB columns to a valid row: their products land only in D columns
    // whose stores are masked below (lanes l and l+16 share the same n).
    const bf16_t* brow = Bm + (long)(nOK ? n : (N - 1)) * K;

    f32x8 acc0 = {};
    f32x8 acc1 = {};
    for (int kb = 0; kb < K; kb += 32) {
        // B fragment: elements 0..7 = K[kb+8h .. +8], 8..15 = K[kb+16+8h .. +8] (shared)
        union { bf16x16 v; bf16x8 h[2]; } bu;
        bu.h[0] = *(const bf16x8*)(brow + kb + 8 * half);
        bu.h[1] = *(const bf16x8*)(brow + kb + 16 + 8 * half);
        bf16x16 b = bu.v;
        // A fragments: elements 0..7 = K[kb+8h .. +8], 8..15 = K[kb+16+8h .. +8]
        union { bf16x16 v; bf16x8 h[2]; } a0, a1;
        a0.h[0] = *(const bf16x8*)(arow0 + kb + 8 * half);
        a0.h[1] = *(const bf16x8*)(arow0 + kb + 16 + 8 * half);
        a1.h[0] = *(const bf16x8*)(arow1 + kb + 8 * half);
        a1.h[1] = *(const bf16x8*)(arow1 + kb + 16 + 8 * half);
        acc0 = __builtin_amdgcn_wmma_f32_16x16x32_bf16(
                   false, a0.v, false, b, (short)0, acc0, false, false);
        acc1 = __builtin_amdgcn_wmma_f32_16x16x32_bf16(
                   false, a1.v, false, b, (short)0, acc1, false, false);
    }

    if (nOK) {
        float bv = bias ? bias[n] : 0.0f;
        #pragma unroll
        for (int r = 0; r < 8; ++r) {
            int row = tp * 32 + r + 8 * half;    // D: element r -> row r + 8*half
            C[(long)row * ldc + n]        = acc0[r] + bv;
            C[(long)(row + 16) * ldc + n] = acc1[r] + bv;
        }
    }
}

// ---------------------------------------------------------------------------
// Helper kernels (f32 VALU; all cheap relative to the GEMMs)
// ---------------------------------------------------------------------------
__global__ void f32_to_bf16_kernel(const float* __restrict__ src,
                                   __hip_bfloat16* __restrict__ dst, int n)
{
    int i = blockIdx.x * blockDim.x + threadIdx.x;
    if (i < n) dst[i] = __float2bfloat16(src[i]);
}

// Fuse w_ih [2048 x 2560] and w_hh [2048 x 512] into wcomb [2048 x 3072] (bf16)
__global__ void build_wcomb_kernel(const float* __restrict__ w_ih,
                                   const float* __restrict__ w_hh,
                                   __hip_bfloat16* __restrict__ wc)
{
    int i = blockIdx.x * blockDim.x + threadIdx.x;
    if (i >= 4 * H_ * KCOMB_) return;
    int j = i / KCOMB_, k = i % KCOMB_;
    float v = (k < E_ + ENC_) ? w_ih[j * (E_ + ENC_) + k]
                              : w_hh[j * H_ + (k - (E_ + ENC_))];
    wc[i] = __float2bfloat16(v);
}

__global__ void add_bias_kernel(const float* __restrict__ a,
                                const float* __restrict__ b,
                                float* __restrict__ o, int n)
{
    int i = blockIdx.x * blockDim.x + threadIdx.x;
    if (i < n) o[i] = a[i] + b[i];
}

// mean over N image features -> meanf [B x ENC]
__global__ void mean_features_kernel(const float* __restrict__ feat,
                                     float* __restrict__ meanf)
{
    int i = blockIdx.x * blockDim.x + threadIdx.x;
    if (i >= B_ * ENC_) return;
    int b = i / ENC_, e = i % ENC_;
    float s = 0.f;
    for (int n = 0; n < N_; ++n) s += feat[((long)(b * N_ + n)) * ENC_ + e];
    meanf[i] = s * (1.0f / (float)N_);
}

// h0 = meanf @ init_h_w.T + b ; c0 likewise. h0 -> bf16 (hbf + abuf slot), c0 -> f32
__global__ void init_state_kernel(const float* __restrict__ meanf,
                                  const float* __restrict__ ihw, const float* __restrict__ ihb,
                                  const float* __restrict__ icw, const float* __restrict__ icb,
                                  float* __restrict__ c,
                                  __hip_bfloat16* __restrict__ hbf,
                                  __hip_bfloat16* __restrict__ abuf)
{
    int i = blockIdx.x * blockDim.x + threadIdx.x;
    if (i >= B_ * H_) return;
    int b = i / H_, j = i % H_;
    const float* m  = meanf + (long)b * ENC_;
    const float* wh = ihw + (long)j * ENC_;
    const float* wc = icw + (long)j * ENC_;
    float sh = 0.f, sc = 0.f;
    for (int e = 0; e < ENC_; ++e) { float mv = m[e]; sh += mv * wh[e]; sc += mv * wc[e]; }
    sh += ihb[j]; sc += icb[j];
    c[i] = sc;
    __hip_bfloat16 hb = __float2bfloat16(sh);
    hbf[i] = hb;
    abuf[(long)b * KCOMB_ + (E_ + ENC_) + j] = hb;
}

// x_t = emb[captions[:, t]] -> bf16 into abuf[:, 0:E]
__global__ void embed_gather_kernel(const int* __restrict__ captions,
                                    const float* __restrict__ emb,
                                    __hip_bfloat16* __restrict__ abuf, int t)
{
    int i = blockIdx.x * blockDim.x + threadIdx.x;
    if (i >= B_ * E_) return;
    int b = i / E_, e = i % E_;
    int tok = captions[b * T_ + t];
    abuf[(long)b * KCOMB_ + e] = __float2bfloat16(emb[(long)tok * E_ + e]);
}

// Bahdanau attention: scores/softmax/context. One block per batch row.
__global__ void __launch_bounds__(128)
attention_kernel(const float* __restrict__ uhs,    // [B, N, ATT]
                 const float* __restrict__ wah,    // [B, ATT]
                 const float* __restrict__ Aw,     // [ATT]
                 const float* __restrict__ Ab,     // [1]
                 const float* __restrict__ feat,   // [B, N, ENC]
                 float* __restrict__ outAlpha,     // + t*N_ already; idx b*STEPS_*N_ + n
                 __hip_bfloat16* __restrict__ abuf)
{
    __shared__ float sc[N_];
    __shared__ float al[N_];
    int b    = blockIdx.x;
    int tid  = threadIdx.x;
    int wv   = tid >> 5;
    int lane = tid & 31;

    const float* wa = wah + (long)b * ATT_;
    for (int n = wv; n < N_; n += 4) {
        const float* u = uhs + ((long)(b * N_ + n)) * ATT_;
        float s = 0.f;
        for (int k = lane; k < ATT_; k += 32)
            s += Aw[k] * tanhf(u[k] + wa[k]);
        for (int off = 16; off; off >>= 1) s += __shfl_xor(s, off, 32);
        if (lane == 0) sc[n] = s + Ab[0];
    }
    __syncthreads();

    if (tid == 0) {
        float mx = sc[0];
        for (int n = 1; n < N_; ++n) mx = fmaxf(mx, sc[n]);
        float sum = 0.f;
        for (int n = 0; n < N_; ++n) { float e = __expf(sc[n] - mx); al[n] = e; sum += e; }
        float inv = 1.f / sum;
        for (int n = 0; n < N_; ++n) al[n] *= inv;
    }
    __syncthreads();

    if (tid < N_) outAlpha[(long)b * (STEPS_ * N_) + tid] = al[tid];

    // context = alpha @ features -> bf16 into abuf[:, E : E+ENC]
    for (int e = tid; e < ENC_; e += 128) {
        float acc = 0.f;
        const float* fb = feat + ((long)b * N_) * ENC_ + e;
        for (int n = 0; n < N_; ++n) acc += al[n] * fb[(long)n * ENC_];
        abuf[(long)b * KCOMB_ + E_ + e] = __float2bfloat16(acc);
    }
}

// LSTM pointwise: c' = sig(f)*c + sig(i)*tanh(g); h' = sig(o)*tanh(c')
__global__ void lstm_pointwise_kernel(const float* __restrict__ gates,
                                      float* __restrict__ c,
                                      __hip_bfloat16* __restrict__ hbf,
                                      __hip_bfloat16* __restrict__ abuf)
{
    int i = blockIdx.x * blockDim.x + threadIdx.x;
    if (i >= B_ * H_) return;
    int b = i / H_, j = i % H_;
    const float* g = gates + (long)b * (4 * H_);
    float ig = g[j], fg = g[H_ + j], gg = g[2 * H_ + j], og = g[3 * H_ + j];
    float si = 1.f / (1.f + __expf(-ig));
    float sf = 1.f / (1.f + __expf(-fg));
    float so = 1.f / (1.f + __expf(-og));
    float c2 = sf * c[i] + si * tanhf(gg);
    float h2 = so * tanhf(c2);
    c[i] = c2;
    __hip_bfloat16 hb = __float2bfloat16(h2);
    hbf[i] = hb;
    abuf[(long)b * KCOMB_ + (E_ + ENC_) + j] = hb;
}

// ---------------------------------------------------------------------------
// Host launcher
// ---------------------------------------------------------------------------
extern "C" void kernel_launch(void* const* d_in, const int* in_sizes, int n_in,
                              void* d_out, int out_size, void* d_ws, size_t ws_size,
                              hipStream_t stream)
{
    (void)in_sizes; (void)n_in; (void)out_size; (void)ws_size;

    const float* features = (const float*)d_in[0];
    const int*   captions = (const int*)  d_in[1];
    const float* emb      = (const float*)d_in[2];
    const float* W_w      = (const float*)d_in[3];
    const float* W_b      = (const float*)d_in[4];
    const float* U_w      = (const float*)d_in[5];
    const float* U_b      = (const float*)d_in[6];
    const float* A_w      = (const float*)d_in[7];
    const float* A_b      = (const float*)d_in[8];
    const float* ihw      = (const float*)d_in[9];
    const float* ihb      = (const float*)d_in[10];
    const float* icw      = (const float*)d_in[11];
    const float* icb      = (const float*)d_in[12];
    const float* w_ih     = (const float*)d_in[13];
    const float* w_hh     = (const float*)d_in[14];
    const float* b_ih     = (const float*)d_in[15];
    const float* b_hh     = (const float*)d_in[16];
    const float* fcn_w    = (const float*)d_in[17];
    const float* fcn_b    = (const float*)d_in[18];
    float* out = (float*)d_out;

    // ---- workspace carve-out (256B aligned) ----
    char*  ws  = (char*)d_ws;
    size_t off = 0;
    auto take = [&](size_t bytes) -> void* {
        void* p = ws + off;
        off = (off + bytes + 255) & ~(size_t)255;
        return p;
    };
    __hip_bfloat16* fcnwb = (__hip_bfloat16*)take((size_t)V_ * H_ * 2);      // fcn_w bf16
    __hip_bfloat16* uwb   = (__hip_bfloat16*)take((size_t)ATT_ * ENC_ * 2);  // U_w bf16
    __hip_bfloat16* wwb   = (__hip_bfloat16*)take((size_t)ATT_ * H_ * 2);    // W_w bf16
    __hip_bfloat16* wcomb = (__hip_bfloat16*)take((size_t)4 * H_ * KCOMB_ * 2);
    __hip_bfloat16* featb = (__hip_bfloat16*)take((size_t)B_ * N_ * ENC_ * 2);
    float*          uhs   = (float*)take((size_t)B_ * N_ * ATT_ * 4);
    float*          meanf = (float*)take((size_t)B_ * ENC_ * 4);
    float*          cst   = (float*)take((size_t)B_ * H_ * 4);               // LSTM c (f32)
    __hip_bfloat16* hbf   = (__hip_bfloat16*)take((size_t)B_ * H_ * 2);      // LSTM h (bf16)
    __hip_bfloat16* abuf  = (__hip_bfloat16*)take((size_t)B_ * KCOMB_ * 2);  // [x|ctx|h] bf16
    float*          wah   = (float*)take((size_t)B_ * ATT_ * 4);
    float*          gates = (float*)take((size_t)B_ * 4 * H_ * 4);
    float*          bcomb = (float*)take((size_t)4 * H_ * 4);

    auto cdiv = [](long a, long b) { return (int)((a + b - 1) / b); };
    auto conv = [&](const float* s, __hip_bfloat16* d, long n) {
        f32_to_bf16_kernel<<<cdiv(n, 256), 256, 0, stream>>>(s, d, (int)n);
    };
    auto gemm = [&](const __hip_bfloat16* A, const __hip_bfloat16* Bm, float* C,
                    const float* bias, int M, int N, int K, long ldc) {
        int waves  = (M / 32) * ((N + 15) / 16);      // one wave per 32x16 block
        int blocks = (waves + 7) / 8;                 // 8 waves / 256-thread block
        gemm_bf16_wmma<<<blocks, 256, 0, stream>>>(A, Bm, C, bias, M, N, K, ldc);
    };

    // ---- one-time precompute (re-done every call: deterministic) ----
    conv(fcn_w, fcnwb, (long)V_ * H_);
    conv(U_w,   uwb,   (long)ATT_ * ENC_);
    conv(W_w,   wwb,   (long)ATT_ * H_);
    conv(features, featb, (long)B_ * N_ * ENC_);
    build_wcomb_kernel<<<cdiv((long)4 * H_ * KCOMB_, 256), 256, 0, stream>>>(w_ih, w_hh, wcomb);
    add_bias_kernel<<<cdiv(4 * H_, 256), 256, 0, stream>>>(b_ih, b_hh, bcomb, 4 * H_);

    mean_features_kernel<<<cdiv((long)B_ * ENC_, 256), 256, 0, stream>>>(features, meanf);
    init_state_kernel<<<cdiv((long)B_ * H_, 256), 256, 0, stream>>>(
        meanf, ihw, ihb, icw, icb, cst, hbf, abuf);

    // u_hs = features @ U_w.T + U_b   : [1568 x 512], K = 2048
    gemm(featb, uwb, uhs, U_b, B_ * N_, ATT_, ENC_, ATT_);

    const size_t PREDTOT = (size_t)B_ * STEPS_ * V_;

    // ---- 20 sequential decode steps ----
    for (int t = 0; t < STEPS_; ++t) {
        embed_gather_kernel<<<cdiv((long)B_ * E_, 256), 256, 0, stream>>>(
            captions, emb, abuf, t);

        // w_ah = h @ W_w.T + W_b : [32 x 512], K = 512
        gemm(hbf, wwb, wah, W_b, B_, ATT_, H_, ATT_);

        attention_kernel<<<B_, 128, 0, stream>>>(
            uhs, wah, A_w, A_b, features,
            out + PREDTOT + (size_t)t * N_, abuf);

        // gates = [x|ctx|h] @ [w_ih|w_hh].T + (b_ih+b_hh) : [32 x 2048], K = 3072
        gemm(abuf, wcomb, gates, bcomb, B_, 4 * H_, KCOMB_, 4 * H_);

        lstm_pointwise_kernel<<<cdiv((long)B_ * H_, 256), 256, 0, stream>>>(
            gates, cst, hbf, abuf);

        // preds[:, t, :] = h' @ fcn_w.T + fcn_b : [32 x 30522], K = 512
        gemm(hbf, fcnwb, out + (size_t)t * V_, fcn_b,
             B_, V_, H_, (long)STEPS_ * V_);
    }
}
